// Attention_74474732913250
// MI455X (gfx1250) — matrix operand order repaired and checked by reading, not compiled, not run on hardware
//
#include <hip/hip_runtime.h>

typedef __attribute__((ext_vector_type(16))) __bf16 v16bf;
typedef __attribute__((ext_vector_type(8)))  __bf16 v8bf;
typedef __attribute__((ext_vector_type(4)))  __bf16 v4bf;
typedef __attribute__((ext_vector_type(8)))  float  v8f;
typedef __attribute__((ext_vector_type(4)))  float  v4f;

#define DEVINL __device__ __forceinline__

// ---------- helpers ----------

DEVINL __bf16 f2bf(float f) {
  union { float f; unsigned u; } x; x.f = f;
  unsigned r = x.u + 0x7FFFu + ((x.u >> 16) & 1u);   // RNE
  unsigned short h = (unsigned short)(r >> 16);
  union { unsigned short s; __bf16 b; } y; y.s = h;
  return y.b;
}

DEVINL v8f wmma_bf16(v16bf a, v16bf b, v8f c) {
  // D = A(16x32) * B(32x16) + C, fp32 accumulate
  return __builtin_amdgcn_wmma_f32_16x16x32_bf16(false, a, false, b, (short)0, c,
                                                 false, false);
}

// A fragment 16x32 bf16. Lane l holds row M = (l&15).
// halves 0..7  = K[kb + (l>>4)*8      .. +7]
// halves 8..15 = K[kb + 16 + (l>>4)*8 .. +7]
DEVINL v16bf a_frag(const __bf16* base, int ld, int row0, int kb, int lane) {
  const __bf16* p = base + (size_t)(row0 + (lane & 15)) * ld + kb + (lane >> 4) * 8;
  v8bf lo = *(const v8bf*)p;
  v8bf hi = *(const v8bf*)(p + 16);
  return __builtin_shufflevector(lo, hi, 0,1,2,3,4,5,6,7,8,9,10,11,12,13,14,15);
}

// B fragment 32x16 bf16. Lane l holds column N = n0 + (l&15),
// halves 0..7 = K[kb + (l>>4)*8 .. +7], halves 8..15 = K[kb + 16 + (l>>4)*8 .. +7].
// "base" is row-major [N][K] (i.e. B transposed: B[k][n] = base[n][k]).
DEVINL v16bf b_frag(const __bf16* base, int ld, int n0, int kb, int lane) {
  const __bf16* p = base + (size_t)(n0 + (lane & 15)) * ld + kb + (lane >> 4) * 8;
  v8bf lo = *(const v8bf*)p;
  v8bf hi = *(const v8bf*)(p + 16);
  return __builtin_shufflevector(lo, hi, 0,1,2,3,4,5,6,7,8,9,10,11,12,13,14,15);
}

// async global -> LDS copy, 16B per lane, tracked by ASYNCcnt.
// Low 32 bits of a generic shared pointer are the workgroup LDS byte offset.
DEVINL void async_g2l_b128(void* lds, const void* g) {
  asm volatile("global_load_async_to_lds_b128 %0, %1, off"
               :: "v"((unsigned)(size_t)lds), "v"(g) : "memory");
}

// ---------- kernel 1: fp32 -> bf16 convert ----------

__global__ __launch_bounds__(256) void k_cvt(const float* __restrict__ in,
                                             __bf16* __restrict__ out, int n4) {
  int i = blockIdx.x * 256 + threadIdx.x;
  if (i >= n4) return;
  v4f f = *(const v4f*)(in + (size_t)i * 4);
  v4bf o;
  #pragma unroll
  for (int e = 0; e < 4; e++) o[e] = f2bf(f[e]);
  *(v4bf*)(out + (size_t)i * 4) = o;
}

// ---------- kernel 2: zero the 196..207 pad rows of q/k/v ----------

__global__ __launch_bounds__(256) void k_init_pads(__bf16* __restrict__ q,
                                                   __bf16* __restrict__ k,
                                                   __bf16* __restrict__ v) {
  int t = blockIdx.x * 256 + threadIdx.x;
  if (t >= 3072 * 768) return;
  int bh = t / 768, rem = t - bh * 768;
  int r = rem >> 6, d = rem & 63;          // 12 pad rows x 64
  __bf16 z = f2bf(0.f);
  size_t idx = (((size_t)bh) * 208 + 196 + r) * 64 + d;
  q[idx] = z; k[idx] = z; v[idx] = z;
}

// ---------- kernel 3: gather rpb into [H,208,208] with -1e30 masking ----------

__global__ __launch_bounds__(256) void k_build_rpb(const float* __restrict__ table,
                                                   const int* __restrict__ rel,
                                                   float* __restrict__ out) {
  int blk = blockIdx.x;            // h*208 + i
  int h = blk / 208, i = blk - h * 208;
  int j = threadIdx.x;
  if (j >= 208) return;
  float v = -1e30f;
  if (i < 196 && j < 196) v = table[rel[i * 196 + j] * 12 + h];
  out[((size_t)h * 208 + i) * 208 + j] = v;
}

// ---------- kernel 4: QKV GEMM [50176,768] x [768,2304] with scatter epilogue ----------

__global__ __launch_bounds__(256) void k_qkv_gemm(const __bf16* __restrict__ xb,
                                                  const __bf16* __restrict__ wb,
                                                  const float* __restrict__ qbias,
                                                  const float* __restrict__ vbias,
                                                  __bf16* __restrict__ qo,
                                                  __bf16* __restrict__ ko,
                                                  __bf16* __restrict__ vo) {
  int lane = threadIdx.x & 31, wv = threadIdx.x >> 5;
  int m0 = blockIdx.y * 128 + (wv & 3) * 32;
  int n0 = blockIdx.x * 128 + (wv >> 2) * 64;
  int ln = lane & 15, hf = lane >> 4;

  v8f acc[2][4];
  #pragma unroll
  for (int i = 0; i < 2; i++)
    #pragma unroll
    for (int j = 0; j < 4; j++) acc[i][j] = (v8f){0,0,0,0,0,0,0,0};

  for (int kb = 0; kb < 768; kb += 32) {
    if (kb + 32 < 768)
      __builtin_prefetch(xb + (size_t)(m0 + ln) * 768 + kb + 32, 0, 3);
    v16bf a0 = a_frag(xb, 768, m0,      kb, lane);
    v16bf a1 = a_frag(xb, 768, m0 + 16, kb, lane);
    v16bf b0 = b_frag(wb, 768, n0,      kb, lane);
    v16bf b1 = b_frag(wb, 768, n0 + 16, kb, lane);
    v16bf b2 = b_frag(wb, 768, n0 + 32, kb, lane);
    v16bf b3 = b_frag(wb, 768, n0 + 48, kb, lane);
    acc[0][0] = wmma_bf16(a0, b0, acc[0][0]);
    acc[0][1] = wmma_bf16(a0, b1, acc[0][1]);
    acc[0][2] = wmma_bf16(a0, b2, acc[0][2]);
    acc[0][3] = wmma_bf16(a0, b3, acc[0][3]);
    acc[1][0] = wmma_bf16(a1, b0, acc[1][0]);
    acc[1][1] = wmma_bf16(a1, b1, acc[1][1]);
    acc[1][2] = wmma_bf16(a1, b2, acc[1][2]);
    acc[1][3] = wmma_bf16(a1, b3, acc[1][3]);
  }

  #pragma unroll
  for (int i = 0; i < 2; i++)
    #pragma unroll
    for (int j = 0; j < 4; j++) {
      int cb = n0 + j * 16;                 // 16-aligned, never straddles 768/64 bounds
      int reg3 = cb / 768;
      int cc = cb - reg3 * 768;
      int h = cc >> 6;
      int d = (cc & 63) + ln;
      float bias = 0.f, scl = 1.f;
      if (reg3 == 0)      { bias = qbias[cc + ln]; scl = 0.125f; }  // hd^-0.5
      else if (reg3 == 2) { bias = vbias[cc + ln]; }
      __bf16* dst = (reg3 == 0) ? qo : (reg3 == 1) ? ko : vo;
      #pragma unroll
      for (int r = 0; r < 8; r++) {
        int m = m0 + i * 16 + r + hf * 8;
        int b = m / 196, n = m - b * 196;
        dst[(((size_t)b * 12 + h) * 208 + n) * 64 + d] =
            f2bf((acc[i][j][r] + bias) * scl);
      }
    }
}

// ---------- kernel 5: attention, one workgroup per (b,h) ----------

__global__ __launch_bounds__(256) void k_attn(const __bf16* __restrict__ qg,
                                              const __bf16* __restrict__ kg,
                                              const __bf16* __restrict__ vg,
                                              const float* __restrict__ rpb,
                                              __bf16* __restrict__ attnout) {
  extern __shared__ __bf16 smem[];
  __bf16* lds_k  = smem;                    // [208][64]
  __bf16* lds_vt = smem + 208 * 64;         // [64][224]  (cols 208..223 zero)
  __bf16* lds_p  = lds_vt + 64 * 224;       // 8 strips of [16][224]

  int bh = blockIdx.x;
  int b = bh / 12, h = bh - b * 12;
  int tid = threadIdx.x;
  int lane = tid & 31, wv = tid >> 5;
  int ln = lane & 15, hf = lane >> 4;

  v8bf z8;
  #pragma unroll
  for (int e = 0; e < 8; e++) z8[e] = f2bf(0.f);

  { // --- stage K via async global->LDS DMA (no VGPR round-trip) ---
    const __bf16* srcK = kg + (size_t)bh * 208 * 64;
    for (int i = tid; i < 1664; i += 256)
      async_g2l_b128(lds_k + i * 8, srcK + i * 8);

    // --- stage V row-major -> LDS transposed [64][224] ---
    const __bf16* srcV = vg + (size_t)bh * 208 * 64;
    for (int i = tid; i < 1664; i += 256) {
      int n = i >> 3, db = (i & 7) * 8;
      v8bf val = *(const v8bf*)(srcV + n * 64 + db);
      #pragma unroll
      for (int e = 0; e < 8; e++) lds_vt[(db + e) * 224 + n] = val[e];
    }
    for (int i = tid; i < 128; i += 256) {  // V^T pad cols 208..223
      int d = i >> 1, c = (i & 1) * 8;
      *(v8bf*)(lds_vt + d * 224 + 208 + c) = z8;
    }
    // zero P-strip pad columns (208..223) once
    int s = tid >> 5, r = (tid >> 1) & 15, c = tid & 1;
    *((v8bf*)(lds_p + (size_t)s * 16 * 224 + r * 224 + 208) + c) = z8;

    asm volatile("s_wait_asynccnt 0" ::: "memory");
  }
  __syncthreads();

  const __bf16* qbase = qg + (size_t)bh * 208 * 64;
  const float* rpbh = rpb + (size_t)h * 208 * 208;
  __bf16* myP = lds_p + (size_t)wv * 16 * 224;

  for (int rt = wv; rt < 13; rt += 8) {
    // S = q_strip(16x64) * K^T(64x208) + rpb   (scale folded into q)
    v16bf qa0 = a_frag(qbase, 64, rt * 16, 0, lane);
    v16bf qa1 = a_frag(qbase, 64, rt * 16, 32, lane);
    v8f S[13];
    #pragma unroll
    for (int nt = 0; nt < 13; nt++) {
      v8f acc = (v8f){0,0,0,0,0,0,0,0};
      acc = wmma_bf16(qa0, b_frag(lds_k, 64, nt * 16, 0, lane), acc);
      acc = wmma_bf16(qa1, b_frag(lds_k, 64, nt * 16, 32, lane), acc);
      const float* rp = rpbh + (size_t)(rt * 16 + hf * 8) * 208 + nt * 16 + ln;
      #pragma unroll
      for (int r = 0; r < 8; r++) acc[r] += rp[r * 208];
      S[nt] = acc;
    }
    // in-register softmax: rows live across 16-lane halves
    float mx[8], sm[8];
    #pragma unroll
    for (int r = 0; r < 8; r++) {
      mx[r] = S[0][r];
      #pragma unroll
      for (int nt = 1; nt < 13; nt++) mx[r] = fmaxf(mx[r], S[nt][r]);
    }
    #pragma unroll
    for (int msk = 1; msk < 16; msk <<= 1)
      #pragma unroll
      for (int r = 0; r < 8; r++) mx[r] = fmaxf(mx[r], __shfl_xor(mx[r], msk, 32));
    #pragma unroll
    for (int r = 0; r < 8; r++) sm[r] = 0.f;
    #pragma unroll
    for (int nt = 0; nt < 13; nt++)
      #pragma unroll
      for (int r = 0; r < 8; r++) {
        float e = __expf(S[nt][r] - mx[r]);
        S[nt][r] = e; sm[r] += e;
      }
    #pragma unroll
    for (int msk = 1; msk < 16; msk <<= 1)
      #pragma unroll
      for (int r = 0; r < 8; r++) sm[r] += __shfl_xor(sm[r], msk, 32);
    #pragma unroll
    for (int r = 0; r < 8; r++) sm[r] = 1.f / sm[r];

    // D-layout -> A-layout through per-wave LDS strip (bf16 P)
    #pragma unroll
    for (int nt = 0; nt < 13; nt++)
      #pragma unroll
      for (int r = 0; r < 8; r++)
        myP[(size_t)(hf * 8 + r) * 224 + nt * 16 + ln] = f2bf(S[nt][r] * sm[r]);
    asm volatile("s_wait_dscnt 0" ::: "memory");   // per-wave LDS RAW fence

    // O = P(16x224) * V(224x64)  (pad region contributes 0)
    v8f O[4];
    #pragma unroll
    for (int j = 0; j < 4; j++) O[j] = (v8f){0,0,0,0,0,0,0,0};
    #pragma unroll
    for (int ks = 0; ks < 7; ks++) {
      v16bf pa = a_frag(myP, 224, 0, ks * 32, lane);
      #pragma unroll
      for (int j = 0; j < 4; j++)
        O[j] = wmma_bf16(pa, b_frag(lds_vt, 224, j * 16, ks * 32, lane), O[j]);
    }
    #pragma unroll
    for (int j = 0; j < 4; j++)
      #pragma unroll
      for (int r = 0; r < 8; r++) {
        int row = rt * 16 + hf * 8 + r;
        if (row < 196)
          attnout[((size_t)b * 196 + row) * 768 + h * 64 + j * 16 + ln] = f2bf(O[j][r]);
      }
  }
}

// ---------- kernel 6: proj GEMM [50176,768] x [768,768] + bias -> fp32 ----------

__global__ __launch_bounds__(256) void k_proj(const __bf16* __restrict__ ab,
                                              const __bf16* __restrict__ wb,
                                              const float* __restrict__ pbias,
                                              float* __restrict__ out) {
  int lane = threadIdx.x & 31, wv = threadIdx.x >> 5;
  int m0 = blockIdx.y * 128 + (wv & 3) * 32;
  int n0 = blockIdx.x * 128 + (wv >> 2) * 64;
  int ln = lane & 15, hf = lane >> 4;

  v8f acc[2][4];
  #pragma unroll
  for (int i = 0; i < 2; i++)
    #pragma unroll
    for (int j = 0; j < 4; j++) acc[i][j] = (v8f){0,0,0,0,0,0,0,0};

  for (int kb = 0; kb < 768; kb += 32) {
    if (kb + 32 < 768)
      __builtin_prefetch(ab + (size_t)(m0 + ln) * 768 + kb + 32, 0, 3);
    v16bf a0 = a_frag(ab, 768, m0,      kb, lane);
    v16bf a1 = a_frag(ab, 768, m0 + 16, kb, lane);
    v16bf b0 = b_frag(wb, 768, n0,      kb, lane);
    v16bf b1 = b_frag(wb, 768, n0 + 16, kb, lane);
    v16bf b2 = b_frag(wb, 768, n0 + 32, kb, lane);
    v16bf b3 = b_frag(wb, 768, n0 + 48, kb, lane);
    acc[0][0] = wmma_bf16(a0, b0, acc[0][0]);
    acc[0][1] = wmma_bf16(a0, b1, acc[0][1]);
    acc[0][2] = wmma_bf16(a0, b2, acc[0][2]);
    acc[0][3] = wmma_bf16(a0, b3, acc[0][3]);
    acc[1][0] = wmma_bf16(a1, b0, acc[1][0]);
    acc[1][1] = wmma_bf16(a1, b1, acc[1][1]);
    acc[1][2] = wmma_bf16(a1, b2, acc[1][2]);
    acc[1][3] = wmma_bf16(a1, b3, acc[1][3]);
  }

  #pragma unroll
  for (int i = 0; i < 2; i++)
    #pragma unroll
    for (int j = 0; j < 4; j++) {
      int col = n0 + j * 16 + ln;
      float pb = pbias[col];
      #pragma unroll
      for (int r = 0; r < 8; r++) {
        int m = m0 + i * 16 + r + hf * 8;
        out[(size_t)m * 768 + col] = acc[i][j][r] + pb;
      }
    }
}

// ---------- host launch ----------

extern "C" void kernel_launch(void* const* d_in, const int* in_sizes, int n_in,
                              void* d_out, int out_size, void* d_ws, size_t ws_size,
                              hipStream_t stream) {
  (void)in_sizes; (void)n_in; (void)out_size; (void)ws_size;
  const float* x      = (const float*)d_in[0];
  const float* qkv_w  = (const float*)d_in[1];
  const float* q_bias = (const float*)d_in[2];
  const float* v_bias = (const float*)d_in[3];
  const float* rpb_t  = (const float*)d_in[4];
  const float* proj_w = (const float*)d_in[5];
  const float* proj_b = (const float*)d_in[6];
  const int*   relidx = (const int*)d_in[7];
  float* out = (float*)d_out;

  char* ws = (char*)d_ws;
  size_t o = 0;
  float*  rpb_pre = (float*)(ws + o);  o += (size_t)12 * 208 * 208 * 4;   //   2.1 MB
  __bf16* xb      = (__bf16*)(ws + o); o += (size_t)50176 * 768 * 2;      //  77.1 MB
  __bf16* wqkv    = (__bf16*)(ws + o); o += (size_t)2304 * 768 * 2;       //   3.5 MB
  __bf16* wproj   = (__bf16*)(ws + o); o += (size_t)768 * 768 * 2;        //   1.2 MB
  __bf16* qb      = (__bf16*)(ws + o); o += (size_t)3072 * 208 * 64 * 2;  //  81.8 MB
  __bf16* kb      = (__bf16*)(ws + o); o += (size_t)3072 * 208 * 64 * 2;  //  81.8 MB
  __bf16* vb      = (__bf16*)(ws + o); o += (size_t)3072 * 208 * 64 * 2;  //  81.8 MB
  __bf16* attno   = (__bf16*)(ws + o); o += (size_t)50176 * 768 * 2;      //  77.1 MB

  hipFuncSetAttribute(reinterpret_cast<const void*>(k_attn),
                      hipFuncAttributeMaxDynamicSharedMemorySize, 112640);

  k_cvt<<<50176 * 768 / 4 / 256, 256, 0, stream>>>(x, xb, 50176 * 768 / 4);
  k_cvt<<<2304 * 768 / 4 / 256, 256, 0, stream>>>(qkv_w, wqkv, 2304 * 768 / 4);
  k_cvt<<<768 * 768 / 4 / 256, 256, 0, stream>>>(proj_w, wproj, 768 * 768 / 4);
  k_init_pads<<<(3072 * 768 + 255) / 256, 256, 0, stream>>>(qb, kb, vb);
  k_build_rpb<<<12 * 208, 256, 0, stream>>>(rpb_t, relidx, rpb_pre);
  k_qkv_gemm<<<dim3(18, 392), 256, 0, stream>>>(xb, wqkv, q_bias, v_bias, qb, kb, vb);
  k_attn<<<3072, 256, 112640, stream>>>(qb, kb, vb, rpb_pre, attno);
  k_proj<<<dim3(6, 392), 256, 0, stream>>>(attno, wproj, proj_b, out);
}
